// BatchRelationalModule_72473278152977
// MI455X (gfx1250) — matrix operand order repaired and checked by reading, not compiled, not run on hardware
//
#include <hip/hip_runtime.h>
#include <hip/hip_bf16.h>

typedef __attribute__((ext_vector_type(16))) __bf16 v16bf;
typedef __attribute__((ext_vector_type(8)))  float  v8f;

#define BATCH 32
#define CIN   128
#define LL    144     // H*W
#define HID   64
#define LSTR  66      // padded LDS row stride: 66x%64==2x (conflict-free), rows 8B-aligned

__device__ __forceinline__ float leaky(float v) { return fmaxf(v, 0.01f * v); }

// ---------------- Kernel 1: zero S accumulator -----------------------------
__global__ void k_zero(float* S, int n) {
    int i = blockIdx.x * blockDim.x + threadIdx.x;
    if (i < n) S[i] = 0.0f;
}

// ---------------- Kernel 2: gi = x@Wa + b1, gj = x@Wb ----------------------
// x[b,l,d] = (d<128) ? x_img[b,d,l] : l ; W1 is (258,64) row-major.
// b1 is folded into gi so the hot loop computes just gi'[j] + gj[i].
__global__ __launch_bounds__(256) void k_gigj(const float* __restrict__ x_img,
                                              const float* __restrict__ W1,
                                              const float* __restrict__ b1,
                                              float* __restrict__ gi,
                                              float* __restrict__ gj) {
    const int b = blockIdx.x;
    const float* xb = x_img + (size_t)b * CIN * LL;
    for (int idx = threadIdx.x; idx < 2 * LL * HID; idx += 256) {
        const int h     = idx & 63;
        const int l     = (idx >> 6) % LL;
        const int which = idx / (LL * HID);          // 0 -> gi (Wa), 1 -> gj (Wb)
        const float* Wbase = W1 + which * 129 * HID;
        float acc = which ? 0.0f : b1[h];
        for (int d = 0; d < CIN; ++d)
            acc += xb[d * LL + l] * Wbase[d * HID + h];
        acc += (float)l * Wbase[CIN * HID + h];      // coord feature (d = 128)
        float* dst = which ? gj : gi;
        dst[((size_t)b * LL + l) * HID + h] = acc;
    }
}

// ---------------- Kernel 3: pairwise WMMA GEMM + reduce --------------------
// For each pair (i,j): h = leaky(gi'[j]+gj[i]);  p = leaky(h@W2 + b2);
// S[b,:] += sum over pairs of p.   leaky(v) = 0.505 v + 0.495 |v| in epilogue.
__global__ __launch_bounds__(256) void k_pair(const float* __restrict__ gi,
                                              const float* __restrict__ gj,
                                              const float* __restrict__ W2,
                                              const float* __restrict__ b2,
                                              float* __restrict__ S) {
    extern __shared__ float smem[];
    float* giL = smem;                       // LL * LSTR
    float* gjL = smem + LL * LSTR;           // LL * LSTR
    float* red = smem + 2 * LL * LSTR;       // 64

    const int b     = blockIdx.x;
    const int chunk = blockIdx.y;
    const int tid   = threadIdx.x;

    // Stage gi'[b], gj[b] into LDS with padded stride.
    const float* giG = gi + (size_t)b * LL * HID;
    const float* gjG = gj + (size_t)b * LL * HID;
    for (int idx = tid; idx < LL * HID; idx += 256) {
        const int l = idx >> 6, h = idx & 63;
        giL[l * LSTR + h] = giG[idx];
        gjL[l * LSTR + h] = gjG[idx];
    }
    if (tid < HID) red[tid] = 0.0f;
    __syncthreads();

    const int wave = tid >> 5;
    const int lane = tid & 31;
    const int half = lane >> 4;   // which 16-lane group
    const int lrow = lane & 15;

    // A fragment (16-bit, 16x32 MxK) per-lane K pairs: element pair p (0..7)
    // starts at K = 32*s + 8*half + (p<4 ? 2p : 2p+8); covers K, K+1.
    // B fragment (16-bit, 32x16 KxN): element e -> K = 32*s + 8*half + (e<8 ? e : e+8), N = lane%16.
    v16bf Bw[2][4];
#pragma unroll
    for (int s = 0; s < 2; ++s)
#pragma unroll
        for (int t = 0; t < 4; ++t)
#pragma unroll
            for (int e = 0; e < 16; ++e) {
                const int k = 32 * s + 8 * half + (e < 8 ? e : e + 8);
                const int n = 16 * t + lrow;
                Bw[s][t][e] = (__bf16)W2[k * HID + n];
            }

    float b2v[4];
#pragma unroll
    for (int t = 0; t < 4; ++t) b2v[t] = b2[16 * t + lrow];

    float s1[4] = {0.0f, 0.0f, 0.0f, 0.0f};   // sum of (c + b2)
    float s2[4] = {0.0f, 0.0f, 0.0f, 0.0f};   // sum of |c + b2|

    // 1296 groups of 16 pairs per batch; group g: i = g/9, j0 = (g%9)*16.
    for (int g = chunk * 8 + wave; g < (LL * LL) / 16; g += gridDim.y * 8) {
        const int i  = g / 9;
        const int j0 = (g % 9) * 16;

        const int irow = i * LSTR;
        const int jrow = (j0 + lrow) * LSTR;

        v16bf a[2];
#pragma unroll
        for (int s = 0; s < 2; ++s)
#pragma unroll
            for (int p = 0; p < 8; ++p) {
                const int kb = 32 * s + 8 * half + (p < 4 ? 2 * p : 2 * p + 8);
                const float2 gv = *(const float2*)&giL[jrow + kb];  // ds_load_b64
                const float2 jv = *(const float2*)&gjL[irow + kb];  // ds_load_b64 (bcast)
                const float f0 = gv.x + jv.x;
                const float f1 = gv.y + jv.y;
                a[s][2 * p]     = (__bf16)leaky(f0);
                a[s][2 * p + 1] = (__bf16)leaky(f1);
            }

#pragma unroll
        for (int t = 0; t < 4; ++t) {
            v8f c = {};
            c = __builtin_amdgcn_wmma_f32_16x16x32_bf16(false, a[0], false, Bw[0][t],
                                                        (short)0, c, false, false);
            c = __builtin_amdgcn_wmma_f32_16x16x32_bf16(false, a[1], false, Bw[1][t],
                                                        (short)0, c, false, false);
#pragma unroll
            for (int r = 0; r < 8; ++r) {
                const float v = c[r] + b2v[t];
                s1[t] += v;
                s2[t] += fabsf(v);   // |.| is a free source modifier
            }
        }
    }

    // leaky(v) = 0.505 v + 0.495 |v|; lane contributes to n = 16*t + lrow.
#pragma unroll
    for (int t = 0; t < 4; ++t)
        atomicAdd(&red[16 * t + lrow], 0.505f * s1[t] + 0.495f * s2[t]);
    __syncthreads();
    if (tid < HID) atomicAdd(&S[b * HID + tid], red[tid]);
}

// ---------------- Kernel 4: two small dense layers -------------------------
__global__ __launch_bounds__(64) void k_head(const float* __restrict__ S,
                                             const float* __restrict__ Wp,
                                             const float* __restrict__ bp,
                                             const float* __restrict__ Wo,
                                             const float* __restrict__ bo,
                                             float* __restrict__ out) {
    __shared__ float tl[HID];
    const int b = blockIdx.x, k = threadIdx.x;
    float acc = bp[k];
    for (int j = 0; j < HID; ++j) acc += S[b * HID + j] * Wp[j * HID + k];
    tl[k] = leaky(acc);
    __syncthreads();
    float acc2 = bo[k];
    for (int j = 0; j < HID; ++j) acc2 += tl[j] * Wo[j * HID + k];
    out[b * HID + k] = leaky(acc2);
}

// ---------------- Launcher -------------------------------------------------
extern "C" void kernel_launch(void* const* d_in, const int* in_sizes, int n_in,
                              void* d_out, int out_size, void* d_ws, size_t ws_size,
                              hipStream_t stream) {
    const float* x_img = (const float*)d_in[0];
    const float* W1    = (const float*)d_in[1];
    const float* b1    = (const float*)d_in[2];
    const float* W2    = (const float*)d_in[3];
    const float* b2    = (const float*)d_in[4];
    const float* Wp    = (const float*)d_in[5];
    const float* bp    = (const float*)d_in[6];
    const float* Wo    = (const float*)d_in[7];
    const float* bo    = (const float*)d_in[8];
    float* out = (float*)d_out;

    float* ws = (float*)d_ws;
    float* gi = ws;                                  // B*L*HID   (holds gi + b1)
    float* gj = ws + (size_t)BATCH * LL * HID;       // B*L*HID
    float* S  = ws + (size_t)2 * BATCH * LL * HID;   // B*HID

    k_zero<<<(BATCH * HID + 255) / 256, 256, 0, stream>>>(S, BATCH * HID);
    k_gigj<<<BATCH, 256, 0, stream>>>(x_img, W1, b1, gi, gj);

    const size_t smem = (size_t)(2 * LL * LSTR + HID) * sizeof(float);  // ~74.5 KB
    dim3 grid(BATCH, 18);
    k_pair<<<grid, 256, smem, stream>>>(gi, gj, W2, b2, S);

    k_head<<<BATCH, HID, 0, stream>>>(S, Wp, bp, Wo, bo, out);
}
